// CrossAttention_11922829213742
// MI455X (gfx1250) — matrix operand order repaired and checked by reading, not compiled, not run on hardware
//
#include <hip/hip_runtime.h>

#define D_   768
#define NH_  6
#define HD_  128
#define FF_  2048
#define S_   256
#define L_   32
#define N_   12
#define C_   12
#define B_   144
#define M_   (B_*S_)   /* 36864 */
#define NL_  2

typedef __attribute__((ext_vector_type(16))) __bf16        v16bf;
typedef __attribute__((ext_vector_type(8)))  float         v8f;
typedef __attribute__((ext_vector_type(4)))  unsigned int  u32x4;

union ABu { v16bf v; u32x4 q[2]; };

__device__ __forceinline__ v8f wmma_bf16(v16bf a, v16bf b, v8f c) {
  return __builtin_amdgcn_wmma_f32_16x16x32_bf16(false, a, false, b, (short)0, c, false, false);
}

__device__ __forceinline__ unsigned short f2bf(float x) {
  unsigned u = __float_as_uint(x);
  u += 0x7fffu + ((u >> 16) & 1u);
  return (unsigned short)(u >> 16);
}

__device__ __forceinline__ v8f vzero8() {
  v8f z = {0.f,0.f,0.f,0.f,0.f,0.f,0.f,0.f};
  return z;
}

// ---------------- elementwise helpers ----------------

__global__ void k_cvt_bf16(const float* __restrict__ s, unsigned short* __restrict__ d, long n) {
  for (long i = blockIdx.x * (long)blockDim.x + threadIdx.x; i < n;
       i += (long)gridDim.x * blockDim.x)
    d[i] = f2bf(s[i]);
}

__global__ void k_bcast(const float* __restrict__ SEQ, float* __restrict__ X,
                        unsigned short* __restrict__ XB) {
  const long total = (long)M_ * D_;
  const long per = (long)S_ * D_;
  for (long i = blockIdx.x * (long)blockDim.x + threadIdx.x; i < total;
       i += (long)gridDim.x * blockDim.x) {
    float v = SEQ[i % per];
    X[i] = v; XB[i] = f2bf(v);
  }
}

// V[b, s, h*HD+d] (stride vStride, base pre-offset to V columns) -> Vt[b, h, d, s]
__global__ void k_vtrans(const unsigned short* __restrict__ V, int vStride,
                         unsigned short* __restrict__ Vt, int keys) {
  const long total = (long)B_ * NH_ * HD_ * keys;
  for (long i = blockIdx.x * (long)blockDim.x + threadIdx.x; i < total;
       i += (long)gridDim.x * blockDim.x) {
    int s = (int)(i % keys); long t = i / keys;
    int d = (int)(t % HD_);  t /= HD_;
    int h = (int)(t % NH_);  int b = (int)(t / NH_);
    Vt[i] = V[(size_t)(b * keys + s) * vStride + h * HD_ + d];
  }
}

// ---------------- generic WMMA GEMM: C[M,N] = A[M,K] * W[N,K]^T + bias --------------
// Block = 4 waves = 64 M rows x 128 N cols. The 128x32 B tile is staged into LDS once
// per k-step with global_load_async_to_lds_b128 (ASYNCcnt), double-buffered, and
// shared by all 4 waves (4x less L2 weight traffic). Optional fused ReLU epilogue.
// B operands are preloaded into registers so each k-step has one dscnt wait, not eight.

#define BROW 40  /* padded LDS row stride in shorts (80B) to spread banks */

template<int OBF, int RELU>
__global__ void __launch_bounds__(128) k_gemm(
    const unsigned short* __restrict__ A, int lda,
    const unsigned short* __restrict__ W,
    const float* __restrict__ bias,
    void* __restrict__ Cp, int ldo,
    int M, int N, int K) {
  __shared__ unsigned short Bt[2][128 * BROW];  // 2 x 10KB
  const int tid = threadIdx.x;
  const int lane = tid & 31;
  const int wv = tid >> 5;
  const int ntiles = N >> 7;
  const int mblk = blockIdx.x / ntiles;
  const int nb = blockIdx.x % ntiles;
  const int n0 = nb << 7;
  const int mt = mblk * 4 + wv;
  const int lrow = lane & 15;
  const int lh = (lane & 16) ? 8 : 0;
  const int bh = (lane & 16) ? 16 : 0;
  const int rb = (lane & 16) ? 8 : 0;

  // thread t stages row t (64B = cols k0..k0+31) of the B tile
  const unsigned short* wsrc = W + (size_t)(n0 + tid) * K;
  const unsigned int lds0 = (unsigned int)(unsigned long long)&Bt[0][tid * BROW];
  const unsigned int lds1 = (unsigned int)(unsigned long long)&Bt[1][tid * BROW];

  v8f acc[8];
#pragma unroll
  for (int t = 0; t < 8; t++) acc[t] = vzero8();
  const unsigned short* Arow = A + (size_t)(mt * 16 + lrow) * lda;

  // prefetch first B tile
  {
    const unsigned short* g = wsrc;
#pragma unroll
    for (int j = 0; j < 4; j++)
      asm volatile("global_load_async_to_lds_b128 %0, %1, off"
                   :: "v"(lds0 + j * 16), "v"(g + j * 8) : "memory");
  }

  for (int k0 = 0; k0 < K; k0 += 32) {
    const int cur = (k0 >> 5) & 1;
    asm volatile("s_wait_asynccnt 0x0" ::: "memory");
    __syncthreads();
    if (k0 + 32 < K) {
      const unsigned short* g = wsrc + k0 + 32;
      const unsigned int l = cur ? lds0 : lds1;
#pragma unroll
      for (int j = 0; j < 4; j++)
        asm volatile("global_load_async_to_lds_b128 %0, %1, off"
                     :: "v"(l + j * 16), "v"(g + j * 8) : "memory");
    }
    ABu a;
    a.q[0] = *reinterpret_cast<const u32x4*>(Arow + k0 + lh);
    a.q[1] = *reinterpret_cast<const u32x4*>(Arow + k0 + 16 + lh);
    const unsigned short* bbase = &Bt[cur][0];
    // preload all 8 B operands (16 ds_load_b128) before the WMMA burst
    ABu bb[8];
#pragma unroll
    for (int t = 0; t < 8; t++) {
      const unsigned short* br = bbase + (t * 16 + lrow) * BROW + lh;
      bb[t].q[0] = *reinterpret_cast<const u32x4*>(br);
      bb[t].q[1] = *reinterpret_cast<const u32x4*>(br + 16);
    }
#pragma unroll
    for (int t = 0; t < 8; t++)
      acc[t] = wmma_bf16(a.v, bb[t].v, acc[t]);
  }
#pragma unroll
  for (int t = 0; t < 8; t++) {
    const int col = n0 + t * 16 + lrow;
    const float bv = bias ? bias[col] : 0.f;
#pragma unroll
    for (int i = 0; i < 8; i++) {
      float v = acc[t][i] + bv;
      if (RELU) v = fmaxf(v, 0.f);
      const size_t idx = (size_t)(mt * 16 + rb + i) * ldo + col;
      if (OBF) ((unsigned short*)Cp)[idx] = f2bf(v);
      else     ((float*)Cp)[idx] = v;
    }
  }
}

// ---------------- flash attention (templated on key count) ----------------

template<int KEYS>
__global__ void __launch_bounds__(128) k_attn(
    const unsigned short* __restrict__ Q, int qStride,
    const unsigned short* __restrict__ Kp, int kStride, int kOff,
    const unsigned short* __restrict__ Vt,
    unsigned short* __restrict__ O, int oStride, float scale) {
  constexpr int NKT = KEYS / 16;
  constexpr int SWM = (KEYS >= 256) ? 7 : 3;
  __shared__ unsigned short Pl[4][16 * KEYS];
  const int lane = threadIdx.x & 31;
  const int wv = threadIdx.x >> 5;
  const int wid = blockIdx.x * 4 + wv;
  const int qt = wid & 15;
  const int h = (wid >> 4) % NH_;
  const int b = wid / (16 * NH_);
  const int lrow = lane & 15;
  const int lh = (lane & 16) ? 8 : 0;
  const int bh = (lane & 16) ? 16 : 0;
  const int rb = (lane & 16) ? 8 : 0;
  unsigned short* Pw = Pl[wv];

  // Q A-operands for 4 k-steps (HD=128)
  ABu qa[4];
  {
    const unsigned short* qrow = Q + (size_t)(b * S_ + qt * 16 + lrow) * qStride + h * HD_;
#pragma unroll
    for (int ks = 0; ks < 4; ks++) {
      qa[ks].q[0] = *reinterpret_cast<const u32x4*>(qrow + ks * 32 + lh);
      qa[ks].q[1] = *reinterpret_cast<const u32x4*>(qrow + ks * 32 + 16 + lh);
    }
  }

  // scores S = Q K^T * scale   (16 x KEYS, kept in registers)
  v8f sc[NKT];
#pragma unroll
  for (int kt = 0; kt < NKT; kt++) {
    v8f c = vzero8();
    const unsigned short* krow = Kp + (size_t)(b * KEYS + kt * 16 + lrow) * kStride + kOff + h * HD_;
#pragma unroll
    for (int ks = 0; ks < 4; ks++) {
      ABu kb;
      kb.q[0] = *reinterpret_cast<const u32x4*>(krow + ks * 32 + lh);
      kb.q[1] = *reinterpret_cast<const u32x4*>(krow + ks * 32 + lh + 16);
      c = wmma_bf16(qa[ks].v, kb.v, c);
    }
#pragma unroll
    for (int i = 0; i < 8; i++) c[i] *= scale;
    sc[kt] = c;
  }

  // row max (C-layout: VGPR i + lane-half <-> row rb+i; 16-lane xor reduce)
  float mx[8], sm[8];
#pragma unroll
  for (int i = 0; i < 8; i++) {
    float m = sc[0][i];
#pragma unroll
    for (int kt = 1; kt < NKT; kt++) m = fmaxf(m, sc[kt][i]);
    mx[i] = m;
  }
#pragma unroll
  for (int d = 1; d < 16; d <<= 1)
#pragma unroll
    for (int i = 0; i < 8; i++) mx[i] = fmaxf(mx[i], __shfl_xor(mx[i], d, 32));

  // exp, sum, store P (bf16) to LDS row-major with bank swizzle
#pragma unroll
  for (int i = 0; i < 8; i++) sm[i] = 0.f;
#pragma unroll
  for (int kt = 0; kt < NKT; kt++) {
#pragma unroll
    for (int i = 0; i < 8; i++) {
      float p = __expf(sc[kt][i] - mx[i]);
      sm[i] += p;
      const int r = rb + i;
      const int col = kt * 16 + lrow;
      Pw[r * KEYS + (col ^ ((r & SWM) << 3))] = f2bf(p);
    }
  }
#pragma unroll
  for (int d = 1; d < 16; d <<= 1)
#pragma unroll
    for (int i = 0; i < 8; i++) sm[i] += __shfl_xor(sm[i], d, 32);
  float rs[8];
#pragma unroll
  for (int i = 0; i < 8; i++) rs[i] = 1.f / sm[i];
  __syncthreads();

  // O = P * V using pre-transposed Vt[b,h,d,keys]
  v8f o[8];
#pragma unroll
  for (int t = 0; t < 8; t++) o[t] = vzero8();
  const int sw = (lrow & SWM) << 3;
#pragma unroll
  for (int kk = 0; kk < KEYS; kk += 32) {
    ABu pa;
    pa.q[0] = *reinterpret_cast<const u32x4*>(&Pw[lrow * KEYS + ((kk + lh) ^ sw)]);
    pa.q[1] = *reinterpret_cast<const u32x4*>(&Pw[lrow * KEYS + ((kk + 16 + lh) ^ sw)]);
#pragma unroll
    for (int t = 0; t < 8; t++) {
      const unsigned short* vr = Vt + ((size_t)((b * NH_ + h) * HD_ + t * 16 + lrow)) * KEYS + kk + lh;
      ABu vb;
      vb.q[0] = *reinterpret_cast<const u32x4*>(vr);
      vb.q[1] = *reinterpret_cast<const u32x4*>(vr + 16);
      o[t] = wmma_bf16(pa.v, vb.v, o[t]);
    }
  }
#pragma unroll
  for (int t = 0; t < 8; t++) {
#pragma unroll
    for (int i = 0; i < 8; i++) {
      const int row = qt * 16 + rb + i;
      const int col = h * HD_ + t * 16 + lrow;
      O[(size_t)(b * S_ + row) * oStride + col] = f2bf(o[t][i] * rs[i]);
    }
  }
}

// ---------------- residual + layernorm (wave per row) ----------------

__global__ void __launch_bounds__(256) k_addln(
    const float* __restrict__ X, const float* __restrict__ Y,
    const float* __restrict__ G, const float* __restrict__ Bt,
    float* __restrict__ Xo, unsigned short* __restrict__ XBo, int rows) {
  const int lane = threadIdx.x & 31;
  const int row = (blockIdx.x * 256 + threadIdx.x) >> 5;
  if (row >= rows) return;
  const float* xr = X + (size_t)row * D_;
  const float* yr = Y + (size_t)row * D_;
  float v[24]; float s = 0.f;
#pragma unroll
  for (int i = 0; i < 24; i++) { v[i] = xr[lane + i * 32] + yr[lane + i * 32]; s += v[i]; }
#pragma unroll
  for (int d = 1; d < 32; d <<= 1) s += __shfl_xor(s, d, 32);
  const float mean = s * (1.f / 768.f);
  float var = 0.f;
#pragma unroll
  for (int i = 0; i < 24; i++) { float t = v[i] - mean; var += t * t; }
#pragma unroll
  for (int d = 1; d < 32; d <<= 1) var += __shfl_xor(var, d, 32);
  const float rstd = rsqrtf(var * (1.f / 768.f) + 1e-5f);
  float* xo = Xo + (size_t)row * D_;
  unsigned short* xb = XBo + (size_t)row * D_;
#pragma unroll
  for (int i = 0; i < 24; i++) {
    const int c = lane + i * 32;
    const float o = (v[i] - mean) * rstd * G[c] + Bt[c];
    xo[c] = o; xb[c] = f2bf(o);
  }
}

// ---------------- small f32 kernels ----------------

__global__ void k_mention(const float* __restrict__ ME, const float* __restrict__ W,
                          const float* __restrict__ Bv, float* __restrict__ out) {
  const int n = blockIdx.x;
  const int o = threadIdx.x;
  if (o >= 300) return;
  float acc = Bv[o];
  const float* m = ME + n * D_;
  const float* w = W + (size_t)o * D_;
  for (int k = 0; k < D_; k++) acc += m[k] * w[k];
  out[n * 300 + o] = acc;
}

__global__ void k_scores0(const float* __restrict__ CDE, const float* __restrict__ men,
                          float* __restrict__ s0) {
  const int nc = blockIdx.x;
  const int n = nc / C_;
  const float* c = CDE + (size_t)nc * 300;
  const float* m = men + n * 300;
  float acc = 0.f;
  for (int o = threadIdx.x; o < 300; o += 32) acc += c[o] * m[o];
  for (int d = 1; d < 32; d <<= 1) acc += __shfl_xor(acc, d, 32);
  if (threadIdx.x == 0) {
    bool valid = (c[0] != 0.f);
    s0[nc] = valid ? acc : -100.f;
  }
}

__global__ void __launch_bounds__(256) k_pool(const float* __restrict__ X,
    const float* __restrict__ CSE, const int* __restrict__ SST, const int* __restrict__ SLN,
    float* __restrict__ nsc) {
  __shared__ float red[256];
  const int nc = blockIdx.x;
  const int n = nc / C_;
  const int st = SST[n];
  const int ln = SLN[n];
  const float inv = 1.f / (float)(ln > 1 ? ln : 1);
  float acc = 0.f;
  for (int d = threadIdx.x; d < D_; d += 256) {
    float pv = 0.f;
    for (int s = 0; s < ln; s++) pv += X[((size_t)nc * S_ + st + s) * D_ + d];
    acc += pv * inv * CSE[(size_t)nc * (L_ * D_) + d];
  }
  red[threadIdx.x] = acc; __syncthreads();
  for (int s = 128; s > 0; s >>= 1) {
    if (threadIdx.x < s) red[threadIdx.x] += red[threadIdx.x + s];
    __syncthreads();
  }
  if (threadIdx.x == 0) nsc[nc] = red[0];
}

__global__ void k_final(const float* __restrict__ nsc, const float* __restrict__ s0,
                        const int* __restrict__ SLN, float* __restrict__ out) {
  const int n = blockIdx.x;
  const int j = threadIdx.x;
  float v = -1e30f;
  if (j < C_) v = (SLN[n] > 0) ? nsc[n * C_ + j] : s0[n * C_ + j];
  else if (j == C_) v = 0.f;
  float m = v;
  for (int d = 1; d < 32; d <<= 1) m = fmaxf(m, __shfl_xor(m, d, 32));
  float e = (j <= C_) ? __expf(v - m) : 0.f;
  float s = e;
  for (int d = 1; d < 32; d <<= 1) s += __shfl_xor(s, d, 32);
  if (j <= C_) out[n * (C_ + 1) + j] = e / s;
}

// ---------------- host orchestration ----------------

extern "C" void kernel_launch(void* const* d_in, const int* in_sizes, int n_in,
                              void* d_out, int out_size, void* d_ws, size_t ws_size,
                              hipStream_t stream) {
  (void)in_sizes; (void)n_in; (void)out_size; (void)ws_size;
  const float* ME   = (const float*)d_in[0];
  const float* SEQ  = (const float*)d_in[1];
  const float* CDE  = (const float*)d_in[2];
  const float* CSE  = (const float*)d_in[3];
  const int*   SST  = (const int*)d_in[4];
  const int*   SLN  = (const int*)d_in[5];
  const float* MPW  = (const float*)d_in[6];
  const float* MPB  = (const float*)d_in[7];
  const float* SAQKVW = (const float*)d_in[8];
  const float* SAQKVB = (const float*)d_in[9];
  const float* SAOUTW = (const float*)d_in[10];
  const float* SAOUTB = (const float*)d_in[11];
  const float* CAQKVW = (const float*)d_in[12];
  const float* CAQKVB = (const float*)d_in[13];
  const float* CAOUTW = (const float*)d_in[14];
  const float* CAOUTB = (const float*)d_in[15];
  const float* FF1W = (const float*)d_in[16];
  const float* FF1B = (const float*)d_in[17];
  const float* FF2W = (const float*)d_in[18];
  const float* FF2B = (const float*)d_in[19];
  const float* N1G = (const float*)d_in[20];
  const float* N1B = (const float*)d_in[21];
  const float* N2G = (const float*)d_in[22];
  const float* N2B = (const float*)d_in[23];
  const float* N3G = (const float*)d_in[24];
  const float* N3B = (const float*)d_in[25];

  char* p = (char*)d_ws;
  auto alloc = [&](size_t bytes) -> char* {
    char* r = p; p += (bytes + 255) & ~(size_t)255; return r;
  };
  float*          X32  = (float*)alloc((size_t)M_ * D_ * 4);
  float*          Y32  = (float*)alloc((size_t)M_ * D_ * 4);
  unsigned short* XB   = (unsigned short*)alloc((size_t)M_ * D_ * 2);
  unsigned short* QKV  = (unsigned short*)alloc((size_t)M_ * 3 * D_ * 2);
  unsigned short* QCA  = (unsigned short*)alloc((size_t)M_ * D_ * 2);
  unsigned short* KVCA = (unsigned short*)alloc((size_t)(B_ * L_) * 2 * D_ * 2);
  unsigned short* VTS  = (unsigned short*)alloc((size_t)B_ * NH_ * HD_ * S_ * 2);
  unsigned short* VTC  = (unsigned short*)alloc((size_t)B_ * NH_ * HD_ * L_ * 2);
  unsigned short* MEMB = (unsigned short*)alloc((size_t)(B_ * L_) * D_ * 2);
  unsigned short* HB   = (unsigned short*)alloc((size_t)M_ * FF_ * 2);  // FFN hidden
  unsigned short* WSAQKV = (unsigned short*)alloc((size_t)NL_ * 3 * D_ * D_ * 2);
  unsigned short* WSAOUT = (unsigned short*)alloc((size_t)NL_ * D_ * D_ * 2);
  unsigned short* WCAQKV = (unsigned short*)alloc((size_t)NL_ * 3 * D_ * D_ * 2);
  unsigned short* WCAOUT = (unsigned short*)alloc((size_t)NL_ * D_ * D_ * 2);
  unsigned short* WFF1   = (unsigned short*)alloc((size_t)NL_ * FF_ * D_ * 2);
  unsigned short* WFF2   = (unsigned short*)alloc((size_t)NL_ * D_ * FF_ * 2);
  float* MEN = (float*)alloc((size_t)N_ * 300 * 4);
  float* SC0 = (float*)alloc((size_t)N_ * C_ * 4);
  float* NSC = (float*)alloc((size_t)N_ * C_ * 4);

  const float scale = 0.08838834764831845f;  // 1/sqrt(HD)

  // weight / input conversions (bf16), deterministic every call
  k_cvt_bf16<<<1024, 256, 0, stream>>>(SAQKVW, WSAQKV, (long)NL_ * 3 * D_ * D_);
  k_cvt_bf16<<<1024, 256, 0, stream>>>(SAOUTW, WSAOUT, (long)NL_ * D_ * D_);
  k_cvt_bf16<<<1024, 256, 0, stream>>>(CAQKVW, WCAQKV, (long)NL_ * 3 * D_ * D_);
  k_cvt_bf16<<<1024, 256, 0, stream>>>(CAOUTW, WCAOUT, (long)NL_ * D_ * D_);
  k_cvt_bf16<<<1024, 256, 0, stream>>>(FF1W, WFF1, (long)NL_ * FF_ * D_);
  k_cvt_bf16<<<1024, 256, 0, stream>>>(FF2W, WFF2, (long)NL_ * D_ * FF_);
  k_cvt_bf16<<<1024, 256, 0, stream>>>(CSE, MEMB, (long)(B_ * L_) * D_);
  k_bcast<<<2048, 256, 0, stream>>>(SEQ, X32, XB);

  // initial candidate scores (tiny)
  k_mention<<<N_, 320, 0, stream>>>(ME, MPW, MPB, MEN);
  k_scores0<<<N_ * C_, 32, 0, stream>>>(CDE, MEN, SC0);

  for (int l = 0; l < NL_; l++) {
    const unsigned short* wsaqkv = WSAQKV + (size_t)l * 3 * D_ * D_;
    const unsigned short* wsaout = WSAOUT + (size_t)l * D_ * D_;
    const unsigned short* wcaqkv = WCAQKV + (size_t)l * 3 * D_ * D_;
    const unsigned short* wcaout = WCAOUT + (size_t)l * D_ * D_;
    const unsigned short* wff1 = WFF1 + (size_t)l * FF_ * D_;
    const unsigned short* wff2 = WFF2 + (size_t)l * D_ * FF_;

    // ---- self attention ----
    // QKV = X Wqkv^T   [M, 3D]
    k_gemm<1,0><<<(M_ / 64) * (3 * D_ / 128), 128, 0, stream>>>(
        XB, D_, wsaqkv, SAQKVB + l * 3 * D_, QKV, 3 * D_, M_, 3 * D_, D_);
    k_vtrans<<<4096, 256, 0, stream>>>(QKV + 2 * D_, 3 * D_, VTS, S_);
    // attention (O overwrites the Q slot of QKV)
    k_attn<S_><<<B_ * NH_ * (S_ / 16) / 4, 128, 0, stream>>>(
        QKV, 3 * D_, QKV, 3 * D_, D_, VTS, QKV, 3 * D_, scale);
    // out projection -> Y32
    k_gemm<0,0><<<(M_ / 64) * (D_ / 128), 128, 0, stream>>>(
        QKV, 3 * D_, wsaout, SAOUTB + l * D_, Y32, D_, M_, D_, D_);
    k_addln<<<(M_ + 7) / 8, 256, 0, stream>>>(X32, Y32, N1G + l * D_, N1B + l * D_, X32, XB, M_);

    // ---- cross attention ----
    k_gemm<1,0><<<(M_ / 64) * (D_ / 128), 128, 0, stream>>>(
        XB, D_, wcaqkv, CAQKVB + l * 3 * D_, QCA, D_, M_, D_, D_);
    k_gemm<1,0><<<((B_ * L_) / 64) * (2 * D_ / 128), 128, 0, stream>>>(
        MEMB, D_, wcaqkv + (size_t)D_ * D_, CAQKVB + l * 3 * D_ + D_,
        KVCA, 2 * D_, B_ * L_, 2 * D_, D_);
    k_vtrans<<<864, 256, 0, stream>>>(KVCA + D_, 2 * D_, VTC, L_);
    k_attn<L_><<<B_ * NH_ * (S_ / 16) / 4, 128, 0, stream>>>(
        QCA, D_, KVCA, 2 * D_, 0, VTC, QCA, D_, scale);
    k_gemm<0,0><<<(M_ / 64) * (D_ / 128), 128, 0, stream>>>(
        QCA, D_, wcaout, CAOUTB + l * D_, Y32, D_, M_, D_, D_);
    k_addln<<<(M_ + 7) / 8, 256, 0, stream>>>(X32, Y32, N2G + l * D_, N2B + l * D_, X32, XB, M_);

    // ---- FFN as two async-staged WMMA GEMMs (H kept bf16, L2-resident) ----
    k_gemm<1,1><<<(M_ / 64) * (FF_ / 128), 128, 0, stream>>>(
        XB, D_, wff1, FF1B + l * FF_, HB, FF_, M_, FF_, D_);
    k_gemm<0,0><<<(M_ / 64) * (D_ / 128), 128, 0, stream>>>(
        HB, FF_, wff2, FF2B + l * D_, Y32, D_, M_, D_, FF_);
    k_addln<<<(M_ + 7) / 8, 256, 0, stream>>>(X32, Y32, N3G + l * D_, N3B + l * D_, X32, XB, M_);
  }

  // span pooling + rescoring + softmax
  k_pool<<<N_ * C_, 256, 0, stream>>>(X32, CSE, SST, SLN, NSC);
  k_final<<<N_, 32, 0, stream>>>(NSC, SC0, SLN, (float*)d_out);
}
